// RNNLM_33595234189382
// MI455X (gfx1250) — matrix operand order repaired and checked by reading, not compiled, not run on hardware
//
#include <hip/hip_runtime.h>

// Problem sizes
#define V_  32000
#define B_  32
#define T_  256
#define E_  128
#define H_  128
#define BT_ 8192      // B*T
#define G3_ 384       // 3*H

typedef __attribute__((ext_vector_type(16))) __bf16 v16bf;
typedef __attribute__((ext_vector_type(8)))  __bf16 v8bf;
typedef __attribute__((ext_vector_type(8)))  float  v8f;

static __device__ __forceinline__ v8f wmma_bf16(v16bf a, v16bf b, v8f c) {
  // D = A(16x32 bf16) * B(32x16 bf16) + C(16x16 f32)
  return __builtin_amdgcn_wmma_f32_16x16x32_bf16(false, a, false, b, (short)0, c,
                                                 false, false);
}

// ---------------------------------------------------------------------------
// A-fragment loader from row-major bf16 [rows][128].
// ISA 16-bit A 16x32 layout: lane = mm + 16*half holds row M=mm; its 16 bf16
// elements are K = kc*32 + {half*8 + 0..7} and K = kc*32 + {16 + half*8 + 0..7}
// -> two contiguous 8-element runs (16B each) in row-major memory.
// ---------------------------------------------------------------------------
static __device__ __forceinline__ v16bf load_a_frag(const __bf16* __restrict__ A,
                                                    int m, int kc, int half) {
  const __bf16* p = A + (size_t)m * 128 + kc * 32 + half * 8;
  v8bf lo = *(const v8bf*)p;         // K run 0 (16B, aligned)
  v8bf hi = *(const v8bf*)(p + 16);  // K run 1
  v16bf a;
#pragma unroll
  for (int e = 0; e < 8; e++) { a[e] = lo[e]; a[8 + e] = hi[e]; }
  return a;
}

// ===========================================================================
// K1: embedding gather.  Writes fp32 emb to d_out (3rd output) and a row-major
// bf16 copy used as the A matrix of the xi GEMM.
// ===========================================================================
__global__ void rnnlm_embed_gather(const int* __restrict__ x,
                                   const float* __restrict__ table,
                                   float* __restrict__ emb_out,
                                   __bf16* __restrict__ embB) {
  for (int idx = blockIdx.x * blockDim.x + threadIdx.x; idx < BT_ * E_;
       idx += gridDim.x * blockDim.x) {
    int m = idx >> 7;          // token position (b*T+t)
    int k = idx & 127;         // embedding dim
    int tok = x[m];
    float v = table[(size_t)tok * E_ + k];
    emb_out[idx] = v;
    embB[idx] = (__bf16)v;
  }
}

// ===========================================================================
// K2: pack a row-major fp32 [K][N] weight into WMMA B-fragment layout (bf16):
// slot = ((nt*KC + kc)*32 + lane)*16 + elem, where lane = n%16 + 16*(kk%16>=8),
// elem = 8*(kk>=16) + kk%8 -> per-lane 32B contiguous fragment loads in GEMMs.
// ===========================================================================
__global__ void rnnlm_pack_b(const float* __restrict__ src,
                             __bf16* __restrict__ dst, int KC, int N, int total) {
  for (int s = blockIdx.x * blockDim.x + threadIdx.x; s < total;
       s += gridDim.x * blockDim.x) {
    int elem = s & 15;
    int lane = (s >> 4) & 31;
    int rest = s >> 9;
    int kc = rest % KC;
    int nt = rest / KC;
    int vg = elem >> 1, pair = elem & 1, half = lane >> 4;
    int n  = nt * 16 + (lane & 15);
    int kk = (elem >> 3) * 16 + half * 8 + (elem & 7);
    int k  = kc * 32 + kk;
    dst[s] = (__bf16)src[(size_t)k * N + n];
  }
}

// ===========================================================================
// K3: xi = emb @ Wi   [8192 x 384], K=128.  One wave per 16x16 tile, 4 WMMAs.
// 12288 waves total.
// ===========================================================================
__global__ __launch_bounds__(256) void rnnlm_xi_gemm(
    const __bf16* __restrict__ embB, const __bf16* __restrict__ WiP,
    float* __restrict__ xi) {
  int tid  = blockIdx.x * 256 + threadIdx.x;
  int wid  = tid >> 5;
  int lane = threadIdx.x & 31;
  int mt = wid / 24;   // 512 M tiles
  int nt = wid % 24;   // 24  N tiles
  int half = lane >> 4, mm = lane & 15;
  int m = mt * 16 + mm;
  v8f c = {};
#pragma unroll
  for (int kc = 0; kc < 4; kc++) {
    v16bf a = load_a_frag(embB, m, kc, half);
    v16bf b = *(const v16bf*)(WiP + (size_t)(((nt * 4 + kc) * 32 + lane)) * 16);
    c = wmma_bf16(a, b, c);
  }
  int row0 = mt * 16 + half * 8;
  int col  = nt * 16 + mm;
#pragma unroll
  for (int i = 0; i < 8; i++) xi[(size_t)(row0 + i) * G3_ + col] = c[i];
}

// ===========================================================================
// K4: GRU scan.  One 1024-thread (32-wave) workgroup, 256 sequential steps.
// Tile->wave assignment is fixed across steps, so Wh B-fragments and bias are
// hoisted into registers: the per-step GEMM phase touches only LDS + WMMA.
// xi_t loads are issued before the GEMM phase to overlap their latency.
// LDS: h (16KB fp32) + gh (48KB fp32) = 64KB.
// ===========================================================================
__global__ __launch_bounds__(1024) void rnnlm_gru_scan(
    const float* __restrict__ xi, const __bf16* __restrict__ WhP,
    const float* __restrict__ bh, const float* __restrict__ carry0,
    __bf16* __restrict__ hsB, float* __restrict__ carry_out) {
  __shared__ float h_lds[B_ * H_];    // 4096 f32
  __shared__ float gh[B_ * G3_];      // 12288 f32
  int tid  = threadIdx.x;
  int lane = tid & 31;
  int w    = __builtin_amdgcn_readfirstlane(tid >> 5);  // wave-uniform wave id
  int half = lane >> 4, mm = lane & 15;

  // Fixed tile assignment: wave w owns tile w, and tile 32+w if w<16 (48 total).
  int mt0 = w / 24, nt0 = w % 24;
  int mt1 = (32 + w) / 24, nt1 = (32 + w) % 24;
  bool has2 = (w < 16);

  // Hoist Wh B-fragments + biases out of the time loop (constant per step).
  v16bf b0[4], b1[4] = {};
#pragma unroll
  for (int kc = 0; kc < 4; kc++)
    b0[kc] = *(const v16bf*)(WhP + (size_t)((nt0 * 4 + kc) * 32 + lane) * 16);
  float bias0 = bh[nt0 * 16 + mm];
  float bias1 = 0.f;
  if (has2) {
#pragma unroll
    for (int kc = 0; kc < 4; kc++)
      b1[kc] = *(const v16bf*)(WhP + (size_t)((nt1 * 4 + kc) * 32 + lane) * 16);
    bias1 = bh[nt1 * 16 + mm];
  }

#pragma unroll
  for (int i = 0; i < 4; i++) h_lds[tid * 4 + i] = carry0[tid * 4 + i];
  __syncthreads();

  int j0 = lane * 4;
  for (int tt = 0; tt < T_; tt++) {
    // ---- issue xi_t loads early: latency overlaps the GEMM phase ----------
    const float* xrow = xi + (size_t)(w * T_ + tt) * G3_;
    float4 xr4 = *(const float4*)(xrow + j0);
    float4 xz4 = *(const float4*)(xrow + 128 + j0);
    float4 xn4 = *(const float4*)(xrow + 256 + j0);

    // ---- GEMM phase: gh = h @ Wh + bh,  48 tiles on 32 waves --------------
    {
      int m = mt0 * 16 + mm;
      v8f c = {};
#pragma unroll
      for (int kc = 0; kc < 4; kc++) {
        const float* p = h_lds + m * 128 + kc * 32 + half * 8;
        float4 l0 = *(const float4*)p;
        float4 l1 = *(const float4*)(p + 4);
        float4 h0 = *(const float4*)(p + 16);
        float4 h1 = *(const float4*)(p + 20);
        float lo[8] = {l0.x, l0.y, l0.z, l0.w, l1.x, l1.y, l1.z, l1.w};
        float hi[8] = {h0.x, h0.y, h0.z, h0.w, h1.x, h1.y, h1.z, h1.w};
        v16bf a;
#pragma unroll
        for (int e = 0; e < 8; e++) { a[e] = (__bf16)lo[e]; a[8 + e] = (__bf16)hi[e]; }
        c = wmma_bf16(a, b0[kc], c);
      }
      int row0 = mt0 * 16 + half * 8;
      int col  = nt0 * 16 + mm;
#pragma unroll
      for (int i = 0; i < 8; i++) gh[(row0 + i) * G3_ + col] = c[i] + bias0;
    }
    if (has2) {  // scalar branch (w is readfirstlane-uniform): EXEC stays all-1s
      int m = mt1 * 16 + mm;
      v8f c = {};
#pragma unroll
      for (int kc = 0; kc < 4; kc++) {
        const float* p = h_lds + m * 128 + kc * 32 + half * 8;
        float4 l0 = *(const float4*)p;
        float4 l1 = *(const float4*)(p + 4);
        float4 h0 = *(const float4*)(p + 16);
        float4 h1 = *(const float4*)(p + 20);
        float lo[8] = {l0.x, l0.y, l0.z, l0.w, l1.x, l1.y, l1.z, l1.w};
        float hi[8] = {h0.x, h0.y, h0.z, h0.w, h1.x, h1.y, h1.z, h1.w};
        v16bf a;
#pragma unroll
        for (int e = 0; e < 8; e++) { a[e] = (__bf16)lo[e]; a[8 + e] = (__bf16)hi[e]; }
        c = wmma_bf16(a, b1[kc], c);
      }
      int row0 = mt1 * 16 + half * 8;
      int col  = nt1 * 16 + mm;
#pragma unroll
      for (int i = 0; i < 8; i++) gh[(row0 + i) * G3_ + col] = c[i] + bias1;
    }
    __syncthreads();

    // ---- elementwise gate phase: 4096 elems, 4 consecutive j per thread ---
    {
      int b = w;  // batch row
      float xr[4] = {xr4.x, xr4.y, xr4.z, xr4.w};
      float xz[4] = {xz4.x, xz4.y, xz4.z, xz4.w};
      float xn[4] = {xn4.x, xn4.y, xn4.z, xn4.w};
      float hnew[4];
#pragma unroll
      for (int i = 0; i < 4; i++) {
        int j = j0 + i;
        float hr = gh[b * G3_ + j];
        float hz = gh[b * G3_ + 128 + j];
        float hn = gh[b * G3_ + 256 + j];
        float ho = h_lds[b * H_ + j];
        float r = 1.f / (1.f + __expf(-(xr[i] + hr)));
        float z = 1.f / (1.f + __expf(-(xz[i] + hz)));
        float n = tanhf(xn[i] + r * hn);
        hnew[i] = (1.f - z) * n + z * ho;
      }
#pragma unroll
      for (int i = 0; i < 4; i++) {
        h_lds[b * H_ + j0 + i] = hnew[i];
        hsB[(size_t)(b * T_ + tt) * H_ + j0 + i] = (__bf16)hnew[i];
      }
    }
    __syncthreads();
  }

#pragma unroll
  for (int i = 0; i < 4; i++) carry_out[tid * 4 + i] = h_lds[tid * 4 + i];
}

// ===========================================================================
// K5: logits = hs @ Wout + bout  [8192 x 32000], K=128.
// Register-blocked: each wave computes 4 M-tiles x 8 N-tiles (128 WMMAs).
// A = 16 fragments held in 128 VGPRs; B streamed once per 64 output rows
// -> packed-Wout L2 traffic drops 4x vs 1-M-tile waves (~1 GB total).
// Nontemporal stores keep the 1.05 GB logits stream out of L2.
// 32000 waves.
// ===========================================================================
__global__ __launch_bounds__(256) void rnnlm_logits_gemm(
    const __bf16* __restrict__ hsB, const __bf16* __restrict__ WoutP,
    const float* __restrict__ bout, float* __restrict__ logits) {
  int tid  = blockIdx.x * 256 + threadIdx.x;
  int wid  = tid >> 5;
  int lane = threadIdx.x & 31;
  const int NG = (V_ / 16) / 8;  // 250 groups of 8 N-tiles
  int mg = wid / NG;             // 128 groups of 4 M-tiles
  int ng = wid % NG;
  int half = lane >> 4, mm = lane & 15;

  v16bf a[4][4];  // [m-sub-tile][kc]
#pragma unroll
  for (int jm = 0; jm < 4; jm++)
#pragma unroll
    for (int kc = 0; kc < 4; kc++)
      a[jm][kc] = load_a_frag(hsB, (mg * 4 + jm) * 16 + mm, kc, half);

#pragma unroll 1
  for (int j = 0; j < 8; j++) {
    int nt = ng * 8 + j;
    const __bf16* bp = WoutP + (size_t)((nt * 4) * 32 + lane) * 16;
    if (j < 7)  // prefetch next N-tile's B fragments (global_prefetch_b8)
      __builtin_prefetch(bp + (size_t)(4 * 32) * 16, 0, 1);
    v16bf b[4];
#pragma unroll
    for (int kc = 0; kc < 4; kc++)
      b[kc] = *(const v16bf*)(bp + (size_t)(kc * 32) * 16);
    v8f c[4] = {{}, {}, {}, {}};
#pragma unroll
    for (int kc = 0; kc < 4; kc++)
#pragma unroll
      for (int jm = 0; jm < 4; jm++)   // 4 independent accumulation chains
        c[jm] = wmma_bf16(a[jm][kc], b[kc], c[jm]);
    int n = nt * 16 + mm;
    float bias = bout[n];
#pragma unroll
    for (int jm = 0; jm < 4; jm++) {
      float* outp = logits + (size_t)((mg * 4 + jm) * 16 + half * 8) * V_ + n;
#pragma unroll
      for (int i = 0; i < 8; i++)
        __builtin_nontemporal_store(c[jm][i] + bias, outp + (size_t)i * V_);
    }
  }
}

// ===========================================================================
// Host-side launcher
// ===========================================================================
extern "C" void kernel_launch(void* const* d_in, const int* in_sizes, int n_in,
                              void* d_out, int out_size, void* d_ws, size_t ws_size,
                              hipStream_t stream) {
  const int*   x      = (const int*)d_in[0];
  const float* table  = (const float*)d_in[1];
  const float* Wi     = (const float*)d_in[2];
  const float* Wh     = (const float*)d_in[3];
  const float* bh     = (const float*)d_in[4];
  const float* Wout   = (const float*)d_in[5];
  const float* bout   = (const float*)d_in[6];
  const float* carry0 = (const float*)d_in[7];

  // Outputs: logits [BT,V], carry [B,H], emb [BT,E] concatenated (fp32).
  float* logits    = (float*)d_out;
  float* carry_out = logits + (size_t)BT_ * V_;
  float* emb_out   = carry_out + (size_t)B_ * H_;

  // Workspace layout (bytes, 256B aligned):
  char* ws = (char*)d_ws;
  __bf16* embB  = (__bf16*)(ws + 0);                    // 2,097,152
  __bf16* hsB   = (__bf16*)(ws + 2097152);              // 2,097,152
  __bf16* WiP   = (__bf16*)(ws + 4194304);              //    98,304
  __bf16* WhP   = (__bf16*)(ws + 4292608);              //    98,304
  __bf16* WoutP = (__bf16*)(ws + 4390912);              // 8,192,000
  float*  xi    = (float*) (ws + 12582912);             // 12,582,912 -> end 24MB

  // K1: embedding gather (+ bf16 copy)
  rnnlm_embed_gather<<<4096, 256, 0, stream>>>(x, table, emb_out, embB);

  // K2: pack B matrices into WMMA fragment layout
  rnnlm_pack_b<<<192, 256, 0, stream>>>(Wi, WiP, 4, G3_, 24 * 4 * 512);
  rnnlm_pack_b<<<192, 256, 0, stream>>>(Wh, WhP, 4, G3_, 24 * 4 * 512);
  rnnlm_pack_b<<<16000, 256, 0, stream>>>(Wout, WoutP, 4, V_, 2000 * 4 * 512);

  // K3: xi = emb @ Wi   (12288 waves)
  rnnlm_xi_gemm<<<1536, 256, 0, stream>>>(embB, WiP, xi);

  // K4: sequential GRU scan (single WGP, 32 waves)
  rnnlm_gru_scan<<<1, 1024, 0, stream>>>(xi, WhP, bh, carry0, hsB, carry_out);

  // K5: logits = hs @ Wout + bout   (32000 register-blocked waves, NT stores)
  rnnlm_logits_gemm<<<4000, 256, 0, stream>>>(hsB, WoutP, bout, logits);
}
